// VectorQuantizer_61521111547967
// MI455X (gfx1250) — matrix operand order repaired and checked by reading, not compiled, not run on hardware
//
#include <hip/hip_runtime.h>
#include <hip/hip_bf16.h>
#include <stdint.h>

typedef __attribute__((ext_vector_type(16))) _Float16 v16h;
typedef __attribute__((ext_vector_type(8)))  _Float16 v8h;
typedef __attribute__((ext_vector_type(8)))  float    v8f;
typedef __attribute__((ext_vector_type(4)))  int      v4i;

#define K_CODES 1024
#define C_DIM   256
#define BATCH   16
#define NPIX    4096          // 64*64
#define ROWS_PER_WAVE 16
#define WAVES_PER_BLOCK 8

#define SLOT_HALVES 24        // 48B slot = 32B payload + 16B pad (bank spread)
#define CHUNK_SLOTS 256       // 8 t-steps * 2 halves * 16 codes

// Async global->LDS path (CDNA5); fall back to ld/st staging if absent.
#if defined(__has_builtin)
#  if __has_builtin(__builtin_amdgcn_global_load_async_to_lds_b128)
#    define VQ_ASYNC 1
#  endif
#endif
#ifndef VQ_ASYNC
#  define VQ_ASYNC 0
#endif

// builtin signature (from hipcc diagnostics): (v4i as1*, v4i as3*, Imm int, Imm int)
#define VQ_GPTR(p) ((__attribute__((address_space(1))) v4i*)(uintptr_t)(p))
#define VQ_LPTR(p) ((__attribute__((address_space(3))) v4i*)(uint32_t)(uintptr_t)(p))

__device__ __forceinline__ void vq_wait_async0() {
#if VQ_ASYNC
#  if __has_builtin(__builtin_amdgcn_s_wait_asynccnt)
    __builtin_amdgcn_s_wait_asynccnt(0);
#  else
    asm volatile("s_wait_asynccnt 0x0" ::: "memory");
#  endif
#endif
}

// -------- workspace layout (bytes) --------
// [0,     4096)  : int   cnt[1024]        (code histogram, zeroed each call)
// [4096,  4100)  : float loss_acc         (zeroed each call)
// [8192, 12288)  : float e_sq[1024]
// [12288,536576) : _Float16 emb16[1024*256]

__global__ __launch_bounds__(256) void vq_prep(const float* __restrict__ emb,
                                               _Float16* __restrict__ emb16,
                                               float* __restrict__ esq,
                                               int* __restrict__ cnt,
                                               float* __restrict__ lossacc) {
    const int k = blockIdx.x;       // codebook row
    const int c = threadIdx.x;      // channel
    float v = emb[k * C_DIM + c];
    emb16[k * C_DIM + c] = (_Float16)v;
    float s = v * v;
    #pragma unroll
    for (int m = 1; m < 32; m <<= 1) s += __shfl_xor(s, m, 32);
    __shared__ float red[WAVES_PER_BLOCK];
    if ((threadIdx.x & 31) == 0) red[threadIdx.x >> 5] = s;
    __syncthreads();
    if (threadIdx.x == 0) {
        float t = 0.f;
        #pragma unroll
        for (int i = 0; i < WAVES_PER_BLOCK; ++i) t += red[i];
        esq[k] = t;
        cnt[k] = 0;
        if (k == 0) *lossacc = 0.f;
    }
}

__global__ __launch_bounds__(256) void vq_main(const float* __restrict__ x,
                                               const float* __restrict__ emb,
                                               const _Float16* __restrict__ emb16,
                                               const float* __restrict__ esq,
                                               int* __restrict__ cnt,
                                               float* __restrict__ lossacc,
                                               float* __restrict__ out_q,
                                               float* __restrict__ out_idx) {
    const int tid  = threadIdx.x;
    const int lane = tid & 31;
    const int w    = tid >> 5;
    const int gw   = blockIdx.x * WAVES_PER_BLOCK + w;   // 0..4095
    const int b    = gw >> 8;                            // 256 waves per batch
    const int n0   = (gw & 255) * ROWS_PER_WAVE;         // pixel tile start
    const int m    = lane & 15;                          // row / column residue
    const int hi   = lane >> 4;                          // half-wave select
    const int hi8  = hi << 3;

    // Double-buffered B staging in fragment order: slot (t,hi,col) holds
    // emb16[chunk*16+col][32t+16hi .. +15] (32B payload, 48B stride).
    __shared__ _Float16 bsm[2][CHUNK_SLOTS * SLOT_HALVES];   // 2 x 12KB

    // cooperative copy: thread tid owns slot tid
    const int t_c   = tid >> 5;
    const int hi_c  = (tid >> 4) & 1;
    const int col_c = tid & 15;
    const _Float16* src0 = emb16 + (size_t)col_c * C_DIM + t_c * 32 + hi_c * 16;

    auto stage = [&](int ch, int buf) {
        const _Float16* s = src0 + (size_t)ch * (16 * C_DIM);
        _Float16* d = &bsm[buf][tid * SLOT_HALVES];
#if VQ_ASYNC
        __builtin_amdgcn_global_load_async_to_lds_b128(VQ_GPTR(s),
                                                       VQ_LPTR(d), 0, 0);
        __builtin_amdgcn_global_load_async_to_lds_b128(VQ_GPTR(s + 8),
                                                       VQ_LPTR(d + 8), 0, 0);
#else
        *(v8h*)d       = *(const v8h*)s;
        *(v8h*)(d + 8) = *(const v8h*)(s + 8);
#endif
    };

    stage(0, 0);   // prefetch first chunk; overlaps A-fragment build

    // ---- build A fragments: 16 rows x 256 channels of x, f32 -> f16 ----
    const float* xb = x + (size_t)b * (C_DIM * NPIX) + n0 + m;
    v16h a[8];
    float xsq = 0.f;
    #pragma unroll
    for (int t = 0; t < 8; ++t) {
        const int kb = 32 * t;
        #pragma unroll
        for (int j = 0; j < 8; ++j) {
            float v1 = xb[(size_t)(kb + 2 * hi8 + j) * NPIX];
            float v2 = xb[(size_t)(kb + 2 * hi8 + 8 + j) * NPIX];
            xsq += v1 * v1 + v2 * v2;
            a[t][j]     = (_Float16)v1;
            a[t][8 + j] = (_Float16)v2;
        }
    }
    xsq += __shfl_xor(xsq, 16, 32);   // full ||x_row||^2

    // ---- sweep all 1024 codes, 16 columns per tile, LDS-shared B ----
    float minv[8];
    int   mini[8];
    #pragma unroll
    for (int r = 0; r < 8; ++r) { minv[r] = 3.0e38f; mini[r] = 0; }

    for (int ch = 0; ch < K_CODES / 16; ++ch) {
        const int buf = ch & 1;
        vq_wait_async0();          // this chunk's async copy has landed
        __syncthreads();           // and everyone is done with the other buffer
        if (ch + 1 < K_CODES / 16) stage(ch + 1, buf ^ 1);

        const int ecol = ch * 16 + m;
        const _Float16* bp = &bsm[buf][(hi * 16 + m) * SLOT_HALVES];
        v8f acc = {};
        #pragma unroll
        for (int t = 0; t < 8; ++t) {
            v8h blo = *(const v8h*)(bp + t * (32 * SLOT_HALVES));
            v8h bhi = *(const v8h*)(bp + t * (32 * SLOT_HALVES) + 8);
            v16h bfrag = __builtin_shufflevector(blo, bhi,
                0, 1, 2, 3, 4, 5, 6, 7, 8, 9, 10, 11, 12, 13, 14, 15);
            acc = __builtin_amdgcn_wmma_f32_16x16x32_f16(
                false, a[t], false, bfrag, (short)0, acc, false, false);
        }
        const float es = esq[ecol];
        #pragma unroll
        for (int r = 0; r < 8; ++r) {
            float d = es - 2.0f * acc[r];   // ||x-e||^2 - ||x||^2
            if (d < minv[r]) { minv[r] = d; mini[r] = ecol; }
        }
    }

    // ---- argmin across the 16 lanes of each half (columns) ----
    #pragma unroll
    for (int r = 0; r < 8; ++r) {
        #pragma unroll
        for (int s = 1; s < 16; s <<= 1) {
            float ov = __shfl_xor(minv[r], s, 32);
            int   oi = __shfl_xor(mini[r], s, 32);
            if (ov < minv[r] || (ov == minv[r] && oi < mini[r])) {
                minv[r] = ov; mini[r] = oi;
            }
        }
    }

    __shared__ float sxsq[WAVES_PER_BLOCK][16];
    __shared__ float sminv[WAVES_PER_BLOCK][16];
    __shared__ int   smini[WAVES_PER_BLOCK][16];
    if (lane < 16) sxsq[w][lane] = xsq;
    if ((lane & 15) == 0) {                 // lane 0 -> rows 0..7, lane 16 -> 8..15
        #pragma unroll
        for (int r = 0; r < 8; ++r) {
            sminv[w][hi8 + r] = minv[r];
            smini[w][hi8 + r] = mini[r];
        }
    }
    __syncthreads();

    // ---- embed_index output + histogram ----
    if (lane < 16) {
        out_idx[(size_t)b * NPIX + n0 + lane] = (float)smini[w][lane];
        atomicAdd(&cnt[smini[w][lane]], 1);
    }
    // ---- commitment-loss partial: sum of min ||x-e||^2 over this tile ----
    if (lane == 0) {
        float s = 0.f;
        #pragma unroll
        for (int r = 0; r < 16; ++r) s += sxsq[w][r] + sminv[w][r];
        atomicAdd(lossacc, s);
    }

    // ---- quantized output: out_q[b,c,n] = embedding[idx_n][c] ----
    const int myidx = smini[w][m];
    const float* erow = emb + (size_t)myidx * C_DIM;
    float* ob = out_q + (size_t)b * (C_DIM * NPIX) + n0 + m;
    for (int c0 = 0; c0 < C_DIM; c0 += 2) {
        const int c = c0 + hi;              // half-waves interleave channels
        ob[(size_t)c * NPIX] = erow[c];     // 16 lanes -> 64B coalesced store
    }
}

__global__ __launch_bounds__(1024) void vq_final(const int* __restrict__ cnt,
                                                 const float* __restrict__ lossacc,
                                                 float* __restrict__ out_loss,
                                                 float* __restrict__ out_usage) {
    const int t = threadIdx.x;
    int z = (cnt[t] == 0) ? 1 : 0;
    #pragma unroll
    for (int m = 1; m < 32; m <<= 1) z += __shfl_xor(z, m, 32);
    __shared__ int red[32];
    if ((t & 31) == 0) red[t >> 5] = z;
    __syncthreads();
    if (t == 0) {
        int s = 0;
        #pragma unroll
        for (int i = 0; i < 32; ++i) s += red[i];
        *out_usage = (float)s / (float)K_CODES;
        *out_loss  = *lossacc * (1.0f / ((float)BATCH * NPIX * C_DIM));
    }
}

extern "C" void kernel_launch(void* const* d_in, const int* in_sizes, int n_in,
                              void* d_out, int out_size, void* d_ws, size_t ws_size,
                              hipStream_t stream) {
    (void)in_sizes; (void)n_in; (void)out_size; (void)ws_size;
    const float* x   = (const float*)d_in[0];
    const float* emb = (const float*)d_in[1];

    char* ws = (char*)d_ws;
    int*      cnt     = (int*)(ws + 0);
    float*    lossacc = (float*)(ws + 4096);
    float*    esq     = (float*)(ws + 8192);
    _Float16* emb16   = (_Float16*)(ws + 12288);

    float* out       = (float*)d_out;
    float* out_q     = out;                                    // 16*256*64*64
    float* out_idx   = out + (size_t)BATCH * C_DIM * NPIX;     // +16777216
    float* out_loss  = out_idx + (size_t)BATCH * NPIX;         // +65536
    float* out_usage = out_loss + 1;

    vq_prep<<<K_CODES, C_DIM, 0, stream>>>(emb, emb16, esq, cnt, lossacc);
    const int nblocks = (BATCH * (NPIX / ROWS_PER_WAVE)) / WAVES_PER_BLOCK; // 512
    vq_main<<<nblocks, 256, 0, stream>>>(x, emb, emb16, esq, cnt, lossacc,
                                         out_q, out_idx);
    vq_final<<<1, 1024, 0, stream>>>(cnt, lossacc, out_loss, out_usage);
}
